// SelfAttention_70858370450005
// MI455X (gfx1250) — compile-verified, matrix operand order checked by reading, not run on hardware
//
#include <hip/hip_runtime.h>
#include <hip/hip_bf16.h>

// ---------------------------------------------------------------------------
// Self-attention for MI455X (gfx1250): bf16 WMMA + async global->LDS copies.
// ---------------------------------------------------------------------------

typedef __attribute__((ext_vector_type(16))) __bf16 v16bf;
typedef __attribute__((ext_vector_type(8)))  float  v8f;
typedef __attribute__((ext_vector_type(4)))  int    i32x4;

#define EMBED     1024
#define HEADS     16
#define HEAD_DIM  64
#define BATCH     4
#define SEQ       2048
#define MTOT      (BATCH * SEQ)      // 8192 tokens

#define BM 128
#define BN 128
#define BK 32
#define KB 64                        // key block in attention kernel

#if __has_builtin(__builtin_amdgcn_global_load_async_to_lds_b128)
#define HAVE_ASYNC_LDS 1
#endif
#if __has_builtin(__builtin_amdgcn_s_wait_asynccnt)
#define HAVE_ASYNC_WAIT 1
#endif

// ----- helpers --------------------------------------------------------------

static __device__ __forceinline__ unsigned short f2bf(float f) {
    unsigned int u = __float_as_uint(f);
    unsigned int r = u + 0x7FFFu + ((u >> 16) & 1u);   // round-to-nearest-even
    return (unsigned short)(r >> 16);
}

static __device__ __forceinline__ __bf16 us2bf(unsigned short u) {
    return __builtin_bit_cast(__bf16, u);
}

// 16-byte global -> LDS copy; async (ASYNCcnt-tracked) on CDNA5 when available.
static __device__ __forceinline__ void async_copy16(void* lds, const void* g) {
#ifdef HAVE_ASYNC_LDS
    __builtin_amdgcn_global_load_async_to_lds_b128(
        (__attribute__((address_space(1))) i32x4*)(void*)g,
        (__attribute__((address_space(3))) i32x4*)lds,
        0, 0);
#else
    *(uint4*)lds = *(const uint4*)g;
#endif
}

template <int N>
static __device__ __forceinline__ void wait_async() {
#ifdef HAVE_ASYNC_WAIT
    __builtin_amdgcn_s_wait_asynccnt(N);
#elif defined(HAVE_ASYNC_LDS)
    asm volatile("s_wait_asynccnt %0" :: "n"(N) : "memory");
#endif
}

// Load a 16x32 bf16 fragment (A layout; also used for B via N-major tiles).
// ISA 16-bit A layout: lanes 0-15 / 16-31 both hold M = lane&15;
// element e maps to K = e + 8*half + (e>=8 ? 8 : 0).
static __device__ __forceinline__ v16bf load_frag(const unsigned short* tile,
                                                  int row, int koff, int ld) {
    const int lane = threadIdx.x & 31;
    const int m    = lane & 15;
    const int half = lane >> 4;
    const unsigned short* p = tile + (row + m) * ld + koff + 8 * half;
    v16bf f;
#pragma unroll
    for (int e = 0; e < 16; ++e) {
        int k = e + ((e >= 8) ? 8 : 0);
        f[e] = us2bf(p[k]);
    }
    return f;
}

static __device__ __forceinline__ v8f wmma_bf16(v16bf a, v16bf b, v8f c) {
    return __builtin_amdgcn_wmma_f32_16x16x32_bf16(
        /*neg_a=*/false, a, /*neg_b=*/false, b,
        /*c_mod=*/(short)0, c, /*reuse_a=*/false, /*reuse_b=*/false);
}

// ----- fp32 -> bf16 conversion ---------------------------------------------

__global__ void __launch_bounds__(256)
convert_kernel(const float* __restrict__ in, unsigned short* __restrict__ out, int n) {
    int i = blockIdx.x * blockDim.x + threadIdx.x;
    const int stride = gridDim.x * blockDim.x;
    for (; i < n; i += stride) out[i] = f2bf(in[i]);
}

// ----- GEMM: C[m,n] = sum_k A[m,k] * W[n,k] + bias[n] ------------------------
// A: [MTOT][EMBED] bf16 row-major.  W: [EMBED][EMBED] bf16 N-major.
// Modes: 0 -> bf16 [B][H][S][Dh], 1 -> bf16 [B][H][Dh][S], 2 -> f32 [M][E].
__global__ void __launch_bounds__(256)
gemm_bf16_kernel(const unsigned short* __restrict__ A,
                 const unsigned short* __restrict__ W,
                 const float* __restrict__ bias,
                 void* __restrict__ out, int mode) {
    __shared__ unsigned short ldsA[2][BM * BK];   // 2 x 8 KB (double buffer)
    __shared__ unsigned short ldsB[2][BN * BK];   // 2 x 8 KB

    const int m0   = blockIdx.x * BM;
    const int n0   = blockIdx.y * BN;
    const int tid  = threadIdx.x;
    const int w    = tid >> 5;           // wave 0..7
    const int wm   = w >> 1;             // 0..3 -> 32-row slab
    const int wn   = w & 1;              // 0..1 -> 64-col slab
    const int lane = tid & 31;
    const int half = lane >> 4;
    const int nc   = lane & 15;

    // staging coordinates: 512 16B-chunks per tile, 2 per thread
    const int row0 = tid >> 2;                 // chunk tid
    const int col0 = (tid & 3) * 8;
    const int row1 = (tid + 256) >> 2;         // chunk tid+256
    const int col1 = ((tid + 256) & 3) * 8;

    v8f acc[2][4];
#pragma unroll
    for (int i = 0; i < 2; ++i)
#pragma unroll
        for (int j = 0; j < 4; ++j)
#pragma unroll
            for (int e = 0; e < 8; ++e) acc[i][j][e] = 0.0f;

    // issue-ahead staging: 4 async ops per thread per k-tile (A0,B0,A1,B1)
    auto stage = [&](int k0, int buf) {
        async_copy16(&ldsA[buf][row0 * BK + col0],
                     &A[(size_t)(m0 + row0) * EMBED + k0 + col0]);
        async_copy16(&ldsB[buf][row0 * BK + col0],
                     &W[(size_t)(n0 + row0) * EMBED + k0 + col0]);
        async_copy16(&ldsA[buf][row1 * BK + col1],
                     &A[(size_t)(m0 + row1) * EMBED + k0 + col1]);
        async_copy16(&ldsB[buf][row1 * BK + col1],
                     &W[(size_t)(n0 + row1) * EMBED + k0 + col1]);
    };

    const int NT = EMBED / BK;    // 32 k-tiles
    stage(0, 0);

    for (int kt = 0; kt < NT; ++kt) {
        const int cur = kt & 1;
        if (kt + 1 < NT) {
            stage((kt + 1) * BK, cur ^ 1);   // prefetch next tile into other buffer
            wait_async<4>();                 // oldest 4 (current tile) complete
        } else {
            wait_async<0>();
        }
        __syncthreads();

        v16bf af[2];
#pragma unroll
        for (int i = 0; i < 2; ++i)
            af[i] = load_frag(&ldsA[cur][0], wm * 32 + i * 16, 0, BK);
#pragma unroll
        for (int j = 0; j < 4; ++j) {
            v16bf bf = load_frag(&ldsB[cur][0], wn * 64 + j * 16, 0, BK);
#pragma unroll
            for (int i = 0; i < 2; ++i)
                acc[i][j] = wmma_bf16(af[i], bf, acc[i][j]);
        }
        __syncthreads();   // protect buffer before it is overwritten next round
    }

    // epilogue (C layout: VGPR r -> row r + 8*half, lane -> col nc)
#pragma unroll
    for (int i = 0; i < 2; ++i) {
#pragma unroll
        for (int j = 0; j < 4; ++j) {
#pragma unroll
            for (int r = 0; r < 8; ++r) {
                const int m = m0 + wm * 32 + i * 16 + r + 8 * half;
                const int n = n0 + wn * 64 + j * 16 + nc;
                const float v = acc[i][j][r] + bias[n];
                if (mode == 2) {
                    ((float*)out)[(size_t)m * EMBED + n] = v;
                } else {
                    const int b = m >> 11, s = m & (SEQ - 1);
                    const int h = n >> 6,  d = n & (HEAD_DIM - 1);
                    const unsigned short bv = f2bf(v);
                    if (mode == 0)
                        ((unsigned short*)out)[(((size_t)(b * HEADS + h) * SEQ) + s) * HEAD_DIM + d] = bv;
                    else
                        ((unsigned short*)out)[(((size_t)(b * HEADS + h) * HEAD_DIM) + d) * SEQ + s] = bv;
                }
            }
        }
    }
}

// ----- Flash-attention-style softmax(QK^T/32) V ------------------------------
// grid = (SEQ/128, BATCH*HEADS), 256 threads (8 waves, 16 query rows each).
__global__ void __launch_bounds__(256)
attention_kernel(const unsigned short* __restrict__ qh,   // [B][H][S][Dh]
                 const unsigned short* __restrict__ kh,   // [B][H][S][Dh]
                 const unsigned short* __restrict__ vt,   // [B][H][Dh][S]
                 unsigned short* __restrict__ ao) {       // [B][S][E]
    __shared__ unsigned short ldsQ[128 * HEAD_DIM];   // 16 KB
    __shared__ unsigned short ldsK[KB * HEAD_DIM];    //  8 KB
    __shared__ unsigned short ldsV[HEAD_DIM * KB];    //  8 KB (V^T block: [d][l])
    __shared__ unsigned short ldsP[8][16 * KB];       // 16 KB (per-wave P staging)

    const int bh   = blockIdx.y;
    const int b    = bh >> 4;
    const int h    = bh & (HEADS - 1);
    const int s0   = blockIdx.x * 128;
    const int tid  = threadIdx.x;
    const int w    = tid >> 5;
    const int lane = tid & 31;
    const int half = lane >> 4;
    const int nc   = lane & 15;

    const unsigned short* qbase = qh + (size_t)(b * HEADS + h) * SEQ * HEAD_DIM;
    const unsigned short* kbase = kh + (size_t)(b * HEADS + h) * SEQ * HEAD_DIM;
    const unsigned short* vbase = vt + (size_t)(b * HEADS + h) * HEAD_DIM * SEQ;

    // stage 128x64 Q block: contiguous 16 KB = 1024 chunks, 4 per thread
#pragma unroll
    for (int i = 0; i < 4; ++i) {
        const int c = tid + i * 256;
        async_copy16(&ldsQ[c * 8], qbase + (size_t)s0 * HEAD_DIM + c * 8);
    }
    wait_async<0>();
    __syncthreads();

    v16bf qf[2];
#pragma unroll
    for (int t = 0; t < 2; ++t) qf[t] = load_frag(ldsQ, w * 16, t * 32, HEAD_DIM);

    v8f oacc[4];
#pragma unroll
    for (int d = 0; d < 4; ++d)
#pragma unroll
        for (int e = 0; e < 8; ++e) oacc[d][e] = 0.0f;

    float run_max[8], run_sum[8];
#pragma unroll
    for (int r = 0; r < 8; ++r) { run_max[r] = -__builtin_inff(); run_sum[r] = 0.0f; }

    const float scale = 0.03125f;   // 1/sqrt(EMBED)

    for (int kb = 0; kb < SEQ / KB; ++kb) {
        // stage K block (contiguous) and V^T block (strided): 2+2 chunks/thread
#pragma unroll
        for (int i = 0; i < 2; ++i) {
            const int c = tid + i * 256;                 // 0..511
            async_copy16(&ldsK[c * 8], kbase + (size_t)kb * KB * HEAD_DIM + c * 8);
            const int row = c >> 3;
            const int col = (c & 7) * 8;
            async_copy16(&ldsV[row * KB + col],
                         &vbase[(size_t)row * SEQ + kb * KB + col]);
        }
        wait_async<0>();
        __syncthreads();

        // scores: S = Q (16x64) x K^T (64x64) for this wave's 16 query rows
        v8f sacc[4];
#pragma unroll
        for (int j = 0; j < 4; ++j)
#pragma unroll
            for (int e = 0; e < 8; ++e) sacc[j][e] = 0.0f;
#pragma unroll
        for (int j = 0; j < 4; ++j) {
#pragma unroll
            for (int t = 0; t < 2; ++t) {
                v16bf kf = load_frag(ldsK, j * 16, t * 32, HEAD_DIM);
                sacc[j] = wmma_bf16(qf[t], kf, sacc[j]);
            }
        }

        // online softmax: row r lives in VGPR r across the 16-lane half-group
#pragma unroll
        for (int r = 0; r < 8; ++r) {
            float lm = sacc[0][r];
#pragma unroll
            for (int j = 1; j < 4; ++j) lm = fmaxf(lm, sacc[j][r]);
#pragma unroll
            for (int mm = 1; mm < 16; mm <<= 1) lm = fmaxf(lm, __shfl_xor(lm, mm, 32));
            lm *= scale;
            const float nm   = fmaxf(run_max[r], lm);
            const float corr = __expf(run_max[r] - nm);
            run_max[r] = nm;

            float psum = 0.0f;
            const int prow = (r + 8 * half) * KB;
#pragma unroll
            for (int j = 0; j < 4; ++j) {
                const float p = __expf(sacc[j][r] * scale - nm);
                psum += p;
                ldsP[w][prow + j * 16 + nc] = f2bf(p);
            }
#pragma unroll
            for (int mm = 1; mm < 16; mm <<= 1) psum += __shfl_xor(psum, mm, 32);
            run_sum[r] = run_sum[r] * corr + psum;
#pragma unroll
            for (int d = 0; d < 4; ++d) oacc[d][r] *= corr;
        }

        // O += P (16x64) x V (64x64); V^T tiles give the B layout directly
#pragma unroll
        for (int t = 0; t < 2; ++t) {
            v16bf pf = load_frag(&ldsP[w][0], 0, t * 32, KB);
#pragma unroll
            for (int d = 0; d < 4; ++d) {
                v16bf vf = load_frag(ldsV, d * 16, t * 32, KB);
                oacc[d] = wmma_bf16(pf, vf, oacc[d]);
            }
        }
        __syncthreads();   // protect ldsK/ldsV before next iteration's staging
    }

    // normalize and scatter bf16 into [B][S][E] at this head's column slice
#pragma unroll
    for (int d = 0; d < 4; ++d) {
#pragma unroll
        for (int r = 0; r < 8; ++r) {
            const int srow = s0 + w * 16 + r + 8 * half;
            const int col  = h * HEAD_DIM + d * 16 + nc;
            const float v  = oacc[d][r] / run_sum[r];
            ao[((size_t)(b * SEQ + srow)) * EMBED + col] = f2bf(v);
        }
    }
}

// ----- launcher --------------------------------------------------------------

extern "C" void kernel_launch(void* const* d_in, const int* in_sizes, int n_in,
                              void* d_out, int out_size, void* d_ws, size_t ws_size,
                              hipStream_t stream) {
    const float* values = (const float*)d_in[0];
    const float* keys   = (const float*)d_in[1];
    const float* query  = (const float*)d_in[2];
    // d_in[3] = mask (reference never applies it)
    const float* Wv = (const float*)d_in[4];
    const float* bv = (const float*)d_in[5];
    const float* Wk = (const float*)d_in[6];
    const float* bk = (const float*)d_in[7];
    const float* Wq = (const float*)d_in[8];
    const float* bq = (const float*)d_in[9];
    const float* Wo = (const float*)d_in[10];
    const float* bo = (const float*)d_in[11];

    char* ws = (char*)d_ws;
    const size_t SZ_ACT = (size_t)MTOT * EMBED * 2;   // 16 MiB per activation
    const size_t SZ_W   = (size_t)EMBED * EMBED * 2;  //  2 MiB per weight

    unsigned short* xq  = (unsigned short*)(ws);
    unsigned short* xk  = (unsigned short*)(ws + SZ_ACT);
    unsigned short* xv  = (unsigned short*)(ws + 2 * SZ_ACT);
    unsigned short* wqb = (unsigned short*)(ws + 3 * SZ_ACT);
    unsigned short* wkb = (unsigned short*)(ws + 3 * SZ_ACT + SZ_W);
    unsigned short* wvb = (unsigned short*)(ws + 3 * SZ_ACT + 2 * SZ_W);
    unsigned short* wob = (unsigned short*)(ws + 3 * SZ_ACT + 3 * SZ_W);
    unsigned short* qh  = (unsigned short*)(ws + 3 * SZ_ACT + 4 * SZ_W);
    unsigned short* kh  = (unsigned short*)(ws + 4 * SZ_ACT + 4 * SZ_W);
    unsigned short* vtb = (unsigned short*)(ws + 5 * SZ_ACT + 4 * SZ_W);
    unsigned short* ao  = xq;   // xq is dead after the Q projection

    const int NACT = MTOT * EMBED;
    const int NW   = EMBED * EMBED;
    dim3 cb(256);

    convert_kernel<<<dim3(2048), cb, 0, stream>>>(query,  xq,  NACT);
    convert_kernel<<<dim3(2048), cb, 0, stream>>>(keys,   xk,  NACT);
    convert_kernel<<<dim3(2048), cb, 0, stream>>>(values, xv,  NACT);
    convert_kernel<<<dim3(512),  cb, 0, stream>>>(Wq, wqb, NW);
    convert_kernel<<<dim3(512),  cb, 0, stream>>>(Wk, wkb, NW);
    convert_kernel<<<dim3(512),  cb, 0, stream>>>(Wv, wvb, NW);
    convert_kernel<<<dim3(512),  cb, 0, stream>>>(Wo, wob, NW);

    dim3 ggrid(MTOT / BM, EMBED / BN);
    gemm_bf16_kernel<<<ggrid, cb, 0, stream>>>(xq, wqb, bq, qh,  0);
    gemm_bf16_kernel<<<ggrid, cb, 0, stream>>>(xk, wkb, bk, kh,  0);
    gemm_bf16_kernel<<<ggrid, cb, 0, stream>>>(xv, wvb, bv, vtb, 1);

    attention_kernel<<<dim3(SEQ / 128, BATCH * HEADS), cb, 0, stream>>>(qh, kh, vtb, ao);

    gemm_bf16_kernel<<<ggrid, cb, 0, stream>>>(ao, wob, bo, d_out, 2);
}
